// NoisyLinear_9809705304417
// MI455X (gfx1250) — matrix operand order repaired and checked by reading, not compiled, not run on hardware
//
#include <hip/hip_runtime.h>

#define IN_F 4096
#define OUT_F 4096
#define BATCH 2048

typedef __attribute__((ext_vector_type(16))) _Float16 v16h;
typedef __attribute__((ext_vector_type(8)))  _Float16 v8h;
typedef __attribute__((ext_vector_type(4)))  _Float16 v4h;
typedef __attribute__((ext_vector_type(8)))  float    v8f;
typedef __attribute__((ext_vector_type(4)))  float    v4f;

#define MT 128
#define NT 128
#define KT 32
#define LDSS 40   // 32 K-halves + 8 pad; 80B row stride (16B multiple -> aligned b128 ds ops)
#define PFD 4     // L2 prefetch distance in K-steps (covers HBM latency beyond the LDS pipeline)

__device__ __forceinline__ float scaled_noise(float e) {
    return copysignf(sqrtf(fabsf(e)), e);   // sign(e)*sqrt(|e|)
}

// Precompute f(eps_in), f(eps_out), fused bias into workspace.
__global__ __launch_bounds__(256) void nl_prep(
    const float* __restrict__ eps_in, const float* __restrict__ eps_out,
    const float* __restrict__ bias_mu, const float* __restrict__ bias_sigma,
    float* __restrict__ fin, float* __restrict__ fout, float* __restrict__ bvec)
{
    int i = blockIdx.x * blockDim.x + threadIdx.x;
    if (i < IN_F)  fin[i] = scaled_noise(eps_in[i]);
    if (i < OUT_F) {
        float f = scaled_noise(eps_out[i]);
        fout[i] = f;
        bvec[i] = bias_mu[i] + bias_sigma[i] * f;
    }
}

// Per-thread prefetch registers for one K-tile (tile rows: lrow + 32*it).
struct Regs {
    v4f xa[4];   // x rows
    v4f mu[4];   // weight_mu rows
    v4f sg[4];   // weight_sigma rows
    v4f fi;      // f(eps_in) slice
};

__device__ __forceinline__ void fetch_tile(
    Regs& R, const float* __restrict__ x, const float* __restrict__ wmu,
    const float* __restrict__ wsg, const float* __restrict__ fin,
    int m0, int n0, int k0, int lrow, int lk)
{
    R.fi = *(const v4f*)&fin[k0 + lk];
    #pragma unroll
    for (int it = 0; it < 4; ++it) {
        const int r = lrow + it * 32;
        R.xa[it] = *(const v4f*)&x  [(size_t)(m0 + r) * IN_F + k0 + lk];
        R.mu[it] = *(const v4f*)&wmu[(size_t)(n0 + r) * IN_F + k0 + lk];
        R.sg[it] = *(const v4f*)&wsg[(size_t)(n0 + r) * IN_F + k0 + lk];
    }
}

__device__ __forceinline__ void stage_tile(
    const Regs& R, const float* __restrict__ fo,
    _Float16* __restrict__ sa, _Float16* __restrict__ sb, int lrow, int lk)
{
    #pragma unroll
    for (int it = 0; it < 4; ++it) {
        const int r = lrow + it * 32;
        v4h ha = { (_Float16)R.xa[it].x, (_Float16)R.xa[it].y,
                   (_Float16)R.xa[it].z, (_Float16)R.xa[it].w };
        *(v4h*)&sa[r * LDSS + lk] = ha;
        const float f0 = fo[it];
        v4h hb = { (_Float16)(R.mu[it].x + R.sg[it].x * (f0 * R.fi.x)),
                   (_Float16)(R.mu[it].y + R.sg[it].y * (f0 * R.fi.y)),
                   (_Float16)(R.mu[it].z + R.sg[it].z * (f0 * R.fi.z)),
                   (_Float16)(R.mu[it].w + R.sg[it].w * (f0 * R.fi.w)) };
        *(v4h*)&sb[r * LDSS + lk] = hb;
    }
}

// out = x @ (mu + sg*fo⊗fi)^T + bvec.
// Block: 256 threads = 8 wave32, block tile 128x128, K-step 32, double-buffered LDS,
// register prefetch one K-step ahead + global_prefetch_b8 PFD K-steps ahead.
// Waves arranged 4(M) x 2(N): each wave owns 32 rows x 64 cols
//   -> 2 A-frags, 4 B-frags, 8 accumulators, 8 WMMAs per K-step, 12 ds_load_b128.
__global__ __launch_bounds__(256) void nl_gemm(
    const float* __restrict__ x,
    const float* __restrict__ wmu,
    const float* __restrict__ wsg,
    const float* __restrict__ fin,
    const float* __restrict__ fout,
    const float* __restrict__ bvec,
    float* __restrict__ out)
{
    __shared__ _Float16 sA[2][MT * LDSS];   // x tile, f16, [row][k]
    __shared__ _Float16 sB[2][NT * LDSS];   // fused weight tile, f16, [col][k]

    const int tid  = threadIdx.x;
    const int lane = tid & 31;
    const int wv   = tid >> 5;
    const int wm   = wv & 3;        // wave M index: rows wm*32..+31
    const int wn   = wv >> 2;       // wave N index: cols wn*64..+63
    const int grp  = lane >> 4;     // half-wave group 0/1
    const int l16  = lane & 15;

    const int m0 = blockIdx.y * MT;
    const int n0 = blockIdx.x * NT;

    // staging map: 8 threads per row, 4 floats (16B) each; 32 rows/pass, 4 passes
    const int lrow = tid >> 3;          // 0..31
    const int lk   = (tid & 7) << 2;    // 0,4,...,28

    // f(eps_out) for this thread's staged weight rows is K-invariant: hoist.
    float fo[4];
    #pragma unroll
    for (int it = 0; it < 4; ++it) fo[it] = fout[n0 + lrow + it * 32];

    v8f acc[2][4];
    #pragma unroll
    for (int ia = 0; ia < 2; ++ia)
        #pragma unroll
        for (int j = 0; j < 4; ++j)
            acc[ia][j] = (v8f){0.f,0.f,0.f,0.f,0.f,0.f,0.f,0.f};

    Regs R;
    fetch_tile(R, x, wmu, wsg, fin, m0, n0, 0, lrow, lk);
    stage_tile(R, fo, sA[0], sB[0], lrow, lk);
    __syncthreads();

    constexpr int NIT = IN_F / KT;      // 128
    int buf = 0;
    for (int itk = 0; itk < NIT; ++itk) {
        const bool hasNext = (itk + 1 < NIT);
        // prefetch next K-tile into registers; latency hides behind the WMMAs
        if (hasNext)
            fetch_tile(R, x, wmu, wsg, fin, m0, n0, (itk + 1) * KT, lrow, lk);

        // HBM->L2 prefetch PFD K-steps ahead. Each row's 32-float K-slice is one
        // 128B cacheline, so only lk==0 lanes issue it (each line touched once).
        {
            int kpf = (itk + PFD) * KT;
            if (kpf > IN_F - KT) kpf = IN_F - KT;
            if (lk == 0) {
                #pragma unroll
                for (int it = 0; it < 4; ++it) {
                    const int r = lrow + it * 32;
                    __builtin_prefetch(&x  [(size_t)(m0 + r) * IN_F + kpf], 0, 0);
                    __builtin_prefetch(&wmu[(size_t)(n0 + r) * IN_F + kpf], 0, 0);
                    __builtin_prefetch(&wsg[(size_t)(n0 + r) * IN_F + kpf], 0, 0);
                }
            }
        }

        const _Float16* sa = sA[buf];
        const _Float16* sb = sB[buf];

        // A fragments (16x32 f16): lanes 0-15 K{0-7,16-23}, lanes 16-31 K{8-15,24-31}
        v16h af[2];
        #pragma unroll
        for (int ia = 0; ia < 2; ++ia) {
            const _Float16* pa = &sa[((wm << 5) + (ia << 4) + l16) * LDSS + (grp << 3)];
            v8h alo = *(const v8h*)pa;
            v8h ahi = *(const v8h*)(pa + 16);
            af[ia] = __builtin_shufflevector(alo, ahi,
                     0,1,2,3,4,5,6,7,8,9,10,11,12,13,14,15);
        }

        // B fragments (32x16): lane=col, lanes 0-15 K{0-7,16-23}, lanes 16-31 K{8-15,24-31}; shared across both A frags
        #pragma unroll
        for (int j = 0; j < 4; ++j) {
            const _Float16* pb = &sb[((wn << 6) + (j << 4) + l16) * LDSS + (grp << 3)];
            v8h blo = *(const v8h*)pb;
            v8h bhi = *(const v8h*)(pb + 16);
            v16h b = __builtin_shufflevector(blo, bhi,
                     0,1,2,3,4,5,6,7,8,9,10,11,12,13,14,15);
            acc[0][j] = __builtin_amdgcn_wmma_f32_16x16x32_f16(
                false, af[0], false, b, (short)0, acc[0][j], false, false);
            acc[1][j] = __builtin_amdgcn_wmma_f32_16x16x32_f16(
                false, af[1], false, b, (short)0, acc[1][j], false, false);
        }

        if (hasNext) {
            stage_tile(R, fo, sA[buf ^ 1], sB[buf ^ 1], lrow, lk);
            __syncthreads();            // single barrier per K-step
        }
        buf ^= 1;
    }

    // epilogue: C layout VGPR v -> row (v + 8*grp), lane%16 -> col; add fused bias
    #pragma unroll
    for (int ia = 0; ia < 2; ++ia) {
        #pragma unroll
        for (int j = 0; j < 4; ++j) {
            const int n  = n0 + (wn << 6) + (j << 4) + l16;
            const float bb = bvec[n];
            const int rbase = m0 + (wm << 5) + (ia << 4) + (grp << 3);
            #pragma unroll
            for (int v = 0; v < 8; ++v)
                out[(size_t)(rbase + v) * OUT_F + n] = acc[ia][j][v] + bb;
        }
    }
}

extern "C" void kernel_launch(void* const* d_in, const int* in_sizes, int n_in,
                              void* d_out, int out_size, void* d_ws, size_t ws_size,
                              hipStream_t stream)
{
    const float* x    = (const float*)d_in[0];
    const float* wmu  = (const float*)d_in[1];
    const float* wsg  = (const float*)d_in[2];
    const float* bmu  = (const float*)d_in[3];
    const float* bsg  = (const float*)d_in[4];
    const float* ein  = (const float*)d_in[5];
    const float* eout = (const float*)d_in[6];
    float* out = (float*)d_out;

    float* fin  = (float*)d_ws;          // IN_F floats
    float* fout = fin + IN_F;            // OUT_F floats
    float* bvec = fout + OUT_F;          // OUT_F floats  (48 KB total)

    nl_prep<<<dim3((IN_F + 255) / 256), dim3(256), 0, stream>>>(
        ein, eout, bmu, bsg, fin, fout, bvec);

    dim3 grid(OUT_F / NT, BATCH / MT);   // 32 x 16 blocks
    nl_gemm<<<grid, dim3(256), 0, stream>>>(x, wmu, wsg, fin, fout, bvec, out);
}
